// FuseNet3_609885356991
// MI455X (gfx1250) — matrix operand order repaired and checked by reading, not compiled, not run on hardware
//
#include <hip/hip_runtime.h>
#include <math.h>

#define BB    8
#define CC    64
#define C2D   128
#define HH    192
#define WD    192
#define HWSZ  (HH*WD)

typedef __attribute__((ext_vector_type(16))) __bf16 v16bf;
typedef __attribute__((ext_vector_type(8)))  float  v8f;

#if defined(__gfx1250__) && __has_builtin(__builtin_amdgcn_tensor_load_to_lds) && __has_builtin(__builtin_amdgcn_s_wait_tensorcnt)
#define HAVE_TDM 1
typedef unsigned int u32x4 __attribute__((ext_vector_type(4)));
typedef int          i32x4 __attribute__((ext_vector_type(4)));
typedef int          i32x8 __attribute__((ext_vector_type(8)));
#endif

union BF16x16 { v16bf v; unsigned short u[16]; unsigned int w[8]; };

__device__ __forceinline__ unsigned short f32_to_bf16(float f) {
  union { float f; unsigned int u; } x; x.f = f;
  if ((x.u & 0x7F800000u) != 0x7F800000u)      // RNE unless inf/nan
    x.u += 0x7FFFu + ((x.u >> 16) & 1u);
  return (unsigned short)(x.u >> 16);
}

// ---- WMMA fragment builders (wave32; ISA 7.12.2 bf16 layouts) -------------
// A (16x32, M=pixels): element (m,k) at base[m*rs + k]; K contiguous -> b32/b128.
__device__ __forceinline__ v16bf load_frag_a(const unsigned short* base, int rs) {
  int lane = threadIdx.x & 31;
  int half = lane >> 4, lr = lane & 15;
  BF16x16 f;
#pragma unroll
  for (int i = 0; i < 8; ++i) {
    int kb = ((i < 4) ? 0 : 16) + half * 8 + (i & 3) * 2;
    f.w[i] = *(const unsigned int*)(base + lr * rs + kb);
  }
  return f.v;
}
// B (32x16, weights pre-interleaved [kpair][n][2]): element (k,n) at
// base[(k>>1)*pitch + n*2 + (k&1)] -> pairs adjacent -> pure b32 loads.
__device__ __forceinline__ v16bf load_frag_b_pairs(const unsigned short* base, int pitch) {
  int lane = threadIdx.x & 31;
  int half = lane >> 4, lr = lane & 15;
  BF16x16 f;
#pragma unroll
  for (int i = 0; i < 8; ++i) {
    int kp = ((i < 4) ? 0 : 8) + half * 4 + (i & 3);
    f.w[i] = *(const unsigned int*)(base + kp * pitch + lr * 2);
  }
  return f.v;
}

// ---------------- setup kernels ----------------
// W = P @ (L @ U); writes bf16 interleaved [ipair][o][2] (B-matrix for 1x1 GEMM)
__global__ void build_w_kernel(const float* __restrict__ l, const float* __restrict__ u,
                               const float* __restrict__ logs, const float* __restrict__ p,
                               const float* __restrict__ sgn,
                               float* __restrict__ lu, unsigned short* __restrict__ wbf) {
  for (int idx = threadIdx.x; idx < C2D*C2D; idx += 256) {
    int i = idx / C2D, j = idx % C2D;
    int kmax = i < j ? i : j;
    float acc = 0.f;
    for (int k = 0; k <= kmax; ++k) {
      float Lik = (k < i) ? l[i*C2D + k] : 1.0f;
      float Ukj = (k < j) ? u[k*C2D + j] : sgn[k] * __expf(logs[k]);
      acc += Lik * Ukj;
    }
    lu[idx] = acc;
  }
  __syncthreads();
  for (int idx = threadIdx.x; idx < C2D*C2D; idx += 256) {
    int o = idx / C2D, j = idx % C2D;
    float acc = 0.f;
    for (int i = 0; i < C2D; ++i) acc += p[o*C2D + i] * lu[i*C2D + j];
    wbf[(j >> 1)*(2*C2D) + o*2 + (j & 1)] = f32_to_bf16(acc);   // B[k=j][n=o]
  }
}

// OIHW (64,64,3,3) -> bf16 [tap][cpair][o][2]
__global__ void conv_w_to_bf16(const float* __restrict__ src, unsigned short* __restrict__ dst) {
  int idx = blockIdx.x * 256 + threadIdx.x;
  if (idx >= CC*9*CC) return;
  int tap = idx / 4096, rem = idx % 4096;
  int cpair = rem >> 7, o = (rem >> 1) & 63, par = rem & 1;
  int c = cpair*2 + par;
  dst[idx] = f32_to_bf16(src[(o*CC + c)*9 + tap]);
}

// [CO][KI] f32 -> bf16 interleaved [kpair][o][2]
__global__ void mat_interleave_bf16(const float* __restrict__ src, unsigned short* __restrict__ dst,
                                    int CO, int KI) {
  int idx = blockIdx.x * 256 + threadIdx.x;
  if (idx >= CO*KI) return;
  int o = idx / KI, i = idx % KI;
  dst[(i >> 1)*(2*CO) + o*2 + (i & 1)] = f32_to_bf16(src[idx]);
}

// NCHW f32 [B][64][HW] -> NHWC bf16 slice of [B*HW][128] at channel offset
__global__ void nchw_to_nhwc_bf16(const float* __restrict__ src, unsigned short* __restrict__ dst,
                                  int choff) {
  __shared__ float tile[64][65];
  int bx = blockIdx.x;
  int pt = bx % (HWSZ/64), b = bx / (HWSZ/64);
  int p0 = pt * 64;
  for (int i = threadIdx.x; i < 64*64; i += 256) {
    int px = i & 63, ch = i >> 6;
    tile[ch][px] = src[((long)b*CC + ch)*HWSZ + p0 + px];     // coalesced read
  }
  __syncthreads();
  for (int i = threadIdx.x; i < 64*64; i += 256) {
    int ch = i & 63, px = i >> 6;
    dst[((long)(b*HWSZ + p0 + px))*C2D + choff + ch] = f32_to_bf16(tile[ch][px]); // coalesced write
  }
}

// NHWC f32 [B*HW][64] -> NCHW f32 [B][64][HW]
__global__ void nhwc_to_nchw_f32(const float* __restrict__ src, float* __restrict__ dst) {
  __shared__ float tile[64][65];
  int bx = blockIdx.x;
  int pt = bx % (HWSZ/64), b = bx / (HWSZ/64);
  int p0 = pt * 64;
  for (int i = threadIdx.x; i < 64*64; i += 256) {
    int ch = i & 63, px = i >> 6;
    tile[px][ch] = src[((long)(b*HWSZ + p0 + px))*CC + ch];   // coalesced read
  }
  __syncthreads();
  for (int i = threadIdx.x; i < 64*64; i += 256) {
    int px = i & 63, ch = i >> 6;
    dst[((long)b*CC + ch)*HWSZ + p0 + px] = tile[px][ch];     // coalesced write
  }
}

// NHWC f32 slice -> NHWC bf16 slice (64 channels)
__global__ void slice_cvt_bf16(const float* __restrict__ src, int spitch,
                               unsigned short* __restrict__ dst, int dpitch) {
  long i = (long)blockIdx.x * 256 + threadIdx.x;
  if (i >= (long)BB*HWSZ*CC) return;
  long p = i >> 6; int c = (int)(i & 63);
  dst[p*dpitch + c] = f32_to_bf16(src[p*spitch + c]);
}

// ---------------- 1x1 conv as WMMA GEMM (NHWC, kin = 128) ----------------
template<int COUT>
__global__ __launch_bounds__(256)
void gemm1x1_wmma(const unsigned short* __restrict__ in_bf,
                  const unsigned short* __restrict__ w_bf,   // [ipair][COUT][2]
                  const float* __restrict__ bias,
                  float* __restrict__ out, long opitch, int mode) {
  constexpr int NITER = COUT / 32;                 // n-tiles per wave
  __shared__ __align__(16) unsigned short in_s[64*136];      // 64 px rows, pitch 136
  __shared__ __align__(16) unsigned short w_s[CC*COUT*2];
  long P0 = (long)blockIdx.x * 64;                 // global pixel
  for (int i = threadIdx.x; i < 64*16; i += 256) { // 64 rows * 16 uint4 (256B/row)
    int row = i >> 4, part = i & 15;
    ((uint4*)in_s)[row*17 + part] = *(const uint4*)(in_bf + (P0 + row)*C2D + part*8);
  }
  for (int i = threadIdx.x; i < CC*COUT*2/8; i += 256)
    ((uint4*)w_s)[i] = ((const uint4*)w_bf)[i];
  __syncthreads();

  int wave = threadIdx.x >> 5, lane = threadIdx.x & 31;
  int half = lane >> 4, lr = lane & 15;
  int mtile = wave & 3, nt0 = wave >> 2;           // pixel tile, first och tile
  v8f z = {0.f,0.f,0.f,0.f,0.f,0.f,0.f,0.f};
  v8f acc[NITER];
#pragma unroll
  for (int j = 0; j < NITER; ++j) acc[j] = z;
#pragma unroll
  for (int k0 = 0; k0 < C2D; k0 += 32) {
    v16bf af = load_frag_a(in_s + (mtile*16)*136 + k0, 136);
#pragma unroll
    for (int j = 0; j < NITER; ++j) {
      int nt = nt0 + 2*j;
      v16bf bf = load_frag_b_pairs(w_s + (k0 >> 1)*(2*COUT) + nt*32, 2*COUT);
      acc[j] = __builtin_amdgcn_wmma_f32_16x16x32_bf16(false, af, false, bf,
                                                       (short)0, acc[j], false, false);
    }
  }
#pragma unroll
  for (int j = 0; j < NITER; ++j) {
    int och = (nt0 + 2*j)*16 + lr;
    float bv = bias ? bias[och] : 0.f;
#pragma unroll
    for (int e = 0; e < 8; ++e) {
      long P = P0 + mtile*16 + e + 8*half;
      float v = acc[j][e] + bv;
      if (mode == 2) v = v > 0.f ? v : 0.01f * v;
      out[P*opitch + och] = v;                     // lanes sweep och: coalesced
    }
  }
}

// ---------------- 3x3 conv, implicit GEMM via WMMA (NHWC) ----------------
// out[P][64] = conv3x3(in)+bias (+res1) (+res2); mode 1: v = 0.8*(2*sigmoid(v)-1)
__global__ __launch_bounds__(256)
void conv3x3_wmma(const unsigned short* __restrict__ in_bf, int in_pitch,
                  const unsigned short* __restrict__ wt_bf,   // [tap][cpair][64][2]
                  const float* __restrict__ bias,
                  const float* __restrict__ res1, int r1p,
                  const float* __restrict__ res2, int r2p,
                  float* __restrict__ out, int mode) {
  __shared__ __align__(16) unsigned short in_s[3*68*68]; // [row][col][ch 0..63 + 4 pad]
  __shared__ __align__(16) unsigned short wt_s[CC*CC];   // one tap: [cpair][o][2]
  int bx = blockIdx.x;
  int xt = bx % 3; int y = (bx / 3) % HH; int b = bx / (3*HH);
  int x0 = xt * 64;
  const unsigned short* inb = in_bf + (long)b * HWSZ * in_pitch;

  bool tdm_done = false;
#ifdef HAVE_TDM
  // Interior tiles: one TDM op fetches the 64ch x 68col x 3row bf16 halo tile.
  // dim0 = channels (memory-contiguous), LDS pad 2 dwords per 32 dwords -> row pitch 68.
  if (xt == 1 && y >= 1 && y <= HH - 2) {
    tdm_done = true;
    if (threadIdx.x < 32) {
      unsigned long long ga =
          (unsigned long long)(const void*)(inb + ((long)(y - 1)*WD + (x0 - 1)) * in_pitch);
      unsigned int lds_addr = (unsigned int)(unsigned long long)(const void*)in_s;
      unsigned int s1 = (unsigned)(WD * in_pitch);           // row stride (elements)
      u32x4 g0;
      g0[0] = 1u;                                  // count = 1
      g0[1] = lds_addr;
      g0[2] = (unsigned int)(ga & 0xFFFFFFFFull);
      g0[3] = (unsigned int)((ga >> 32) & 0x1FFFFFFull) | (2u << 30);  // type=2
      i32x8 g1;
      g1[0] = (int)((1u << 16) | (1u << 20) | (4u << 22) | (1u << 25));
      //        data_size=2B    pad_enable   interval=32dw  amount=2dw
      g1[1] = (int)(64u << 16);                    // tensor_dim0 lo16 (=64 ch)
      g1[2] = (int)((unsigned)WD << 16);           // td0 hi | tensor_dim1 lo16
      g1[3] = (int)(64u << 16);                    // td1 hi | tile_dim0 = 64
      g1[4] = (int)(68u | (3u << 16));             // tile_dim1 = 68, tile_dim2 = 3
      g1[5] = (int)in_pitch;                       // tensor_dim0_stride lo32
      g1[6] = (int)(s1 << 16);                     // s0 hi | tensor_dim1_stride lo16
      g1[7] = (int)(s1 >> 16);                     // tensor_dim1_stride[47:16]
      i32x4 g2; g2[0] = HH; g2[1] = 0; g2[2] = 0; g2[3] = 0;   // tensor_dim2
      i32x4 g3; g3[0] = 0;  g3[1] = 0; g3[2] = 0; g3[3] = 0;
#if __clang_major__ >= 23
      i32x8 gz; for (int q = 0; q < 8; ++q) gz[q] = 0;
      __builtin_amdgcn_tensor_load_to_lds(g0, g1, g2, g3, gz, 0);
#else
      __builtin_amdgcn_tensor_load_to_lds(g0, g1, g2, g3, 0);
#endif
      __builtin_amdgcn_s_wait_tensorcnt(0);
    }
  }
#endif
  if (!tdm_done) {
    for (int i = threadIdx.x; i < 3*68*68; i += 256) {
      int c = i % 68, col = (i / 68) % 68, r = i / (68*68);
      int yy = y + r - 1, xx = x0 + col - 1;
      unsigned short v = 0;
      if (c < 64 && col < 66 && yy >= 0 && yy < HH && xx >= 0 && xx < WD)
        v = inb[((long)yy*WD + xx)*in_pitch + c];              // coalesced (c fast)
      in_s[i] = v;
    }
  }

  int wave = threadIdx.x >> 5, lane = threadIdx.x & 31;
  int half = lane >> 4, lr = lane & 15;
  int mtile = wave >> 1;             // pixel tile 0..3
  int ntile0 = (wave & 1) * 2;       // two out-channel tiles
  v8f acc0 = {0.f,0.f,0.f,0.f,0.f,0.f,0.f,0.f};
  v8f acc1 = {0.f,0.f,0.f,0.f,0.f,0.f,0.f,0.f};

  for (int tap = 0; tap < 9; ++tap) {
    __syncthreads();
    for (int i = threadIdx.x; i < CC*CC/8; i += 256) {      // stage 8KB weight slice
      ((uint4*)wt_s)[i] = ((const uint4*)(wt_bf + (long)tap*4096))[i];
    }
    __syncthreads();
    if (tap < 8)   // prefetch next tap slice (global_prefetch_b8)
      __builtin_prefetch(wt_bf + (long)(tap + 1)*4096 + (threadIdx.x & 63)*64, 0, 1);
    int dy = tap / 3, dx = tap % 3;
#pragma unroll
    for (int c0 = 0; c0 < CC; c0 += 32) {
      v16bf af = load_frag_a(in_s + (dy*68 + dx + mtile*16)*68 + c0, 68);
      v16bf b0 = load_frag_b_pairs(wt_s + (c0 >> 1)*128 + ntile0*32, 128);
      v16bf b1 = load_frag_b_pairs(wt_s + (c0 >> 1)*128 + (ntile0+1)*32, 128);
      acc0 = __builtin_amdgcn_wmma_f32_16x16x32_bf16(false, af, false, b0, (short)0, acc0, false, false);
      acc1 = __builtin_amdgcn_wmma_f32_16x16x32_bf16(false, af, false, b1, (short)0, acc1, false, false);
    }
  }
#pragma unroll
  for (int t = 0; t < 2; ++t) {
    v8f acc = t ? acc1 : acc0;
    int och = (ntile0 + t)*16 + lr;
    float bv = bias[och];
#pragma unroll
    for (int e = 0; e < 8; ++e) {
      long P = (long)b*HWSZ + (long)y*WD + x0 + mtile*16 + e + 8*half;
      float v = acc[e] + bv;
      if (res1) v += res1[P*r1p + och];
      if (res2) v += res2[P*r2p + och];
      if (mode == 1) v = 0.8f * (2.f / (1.f + __expf(-v)) - 1.f);
      out[P*64 + och] = v;                          // lanes sweep och: coalesced
    }
  }
}

// ---------------- HIN stats (two-stage, deterministic) / normalize ----------------
__global__ void hin_stats1(const float* __restrict__ r, float* __restrict__ part) {
  int blk = blockIdx.x;                 // BB*16
  int b = blk >> 4, ck = blk & 15;
  int c = threadIdx.x & 63, pg = threadIdx.x >> 6;
  const float* pr = r + ((long)b*HWSZ + (long)ck*(HWSZ/16))*64 + c;
  float s = 0.f, s2 = 0.f;
  for (int j = pg; j < HWSZ/16; j += 4) {           // coalesced (c fast across lanes)
    float v = pr[(long)j*64];
    s += v; s2 += v*v;
  }
  __shared__ float sh[256], sh2[256];
  sh[threadIdx.x] = s; sh2[threadIdx.x] = s2; __syncthreads();
  if (threadIdx.x < 64) {
    float ts = sh[threadIdx.x] + sh[threadIdx.x+64] + sh[threadIdx.x+128] + sh[threadIdx.x+192];
    float t2 = sh2[threadIdx.x] + sh2[threadIdx.x+64] + sh2[threadIdx.x+128] + sh2[threadIdx.x+192];
    part[(long)blk*128 + threadIdx.x*2]     = ts;
    part[(long)blk*128 + threadIdx.x*2 + 1] = t2;
  }
}

__global__ void hin_stats2(const float* __restrict__ part, float* __restrict__ stats) {
  int t = threadIdx.x;                  // 256 = BB*32
  if (t >= BB*32) return;
  int b = t >> 5, c = t & 31;
  float s = 0.f, s2 = 0.f;
  for (int k = 0; k < 16; ++k) {
    s  += part[(long)(b*16 + k)*128 + c*2];
    s2 += part[(long)(b*16 + k)*128 + c*2 + 1];
  }
  float mean = s / (float)HWSZ;
  float var  = s2 / (float)HWSZ - mean*mean;
  stats[t*2]     = mean;
  stats[t*2 + 1] = rsqrtf(var + 1e-5f);
}

__global__ void hin_norm_cvt(const float* __restrict__ r, const float* __restrict__ stats,
                             const float* __restrict__ gamma, const float* __restrict__ beta,
                             unsigned short* __restrict__ dst) {
  long i = (long)blockIdx.x * 256 + threadIdx.x;
  if (i >= (long)BB*HWSZ*CC) return;
  long p = i >> 6; int c = (int)(i & 63); int b = (int)(p / HWSZ);
  float v = r[i];
  if (c < 32) {
    float mean = stats[(b*32 + c)*2], rstd = stats[(b*32 + c)*2 + 1];
    v = (v - mean) * rstd * gamma[c] + beta[c];
  }
  dst[i] = f32_to_bf16(v);
}

// y2 = x2*exp(s) + g -> bf16 into ycat[:,64:128]  (all NHWC)
__global__ void y2_kernel(const float* __restrict__ x2 /*[P][128]+64*/, const float* __restrict__ s,
                          const float* __restrict__ g, unsigned short* __restrict__ ycat2) {
  long i = (long)blockIdx.x * 256 + threadIdx.x;
  if (i >= (long)BB*HWSZ*CC) return;
  long p = i >> 6; int c = (int)(i & 63);
  float v = x2[p*C2D + c] * __expf(s[i]) + g[i];
  ycat2[p*C2D + c] = f32_to_bf16(v);
}

// pooled[b,c] = max + mean over HW (reads NCHW d_out)
__global__ void pool_kernel(const float* __restrict__ x, float* __restrict__ pooled) {
  int bc = blockIdx.x;                      // B*64
  const float* p = x + (long)bc * HWSZ;
  float mx = -3.402823466e38f, s = 0.f;
  for (int i = threadIdx.x; i < HWSZ; i += 256) { float v = p[i]; mx = fmaxf(mx, v); s += v; }
  __shared__ float shm[256], shs[256];
  shm[threadIdx.x] = mx; shs[threadIdx.x] = s; __syncthreads();
  for (int st = 128; st > 0; st >>= 1) {
    if (threadIdx.x < st) { shm[threadIdx.x] = fmaxf(shm[threadIdx.x], shm[threadIdx.x+st]); shs[threadIdx.x] += shs[threadIdx.x+st]; }
    __syncthreads();
  }
  if (threadIdx.x == 0) pooled[bc] = shm[0] + shs[0] / (float)HWSZ;
}

__global__ void gate_kernel(const float* __restrict__ pooled,
                            const float* __restrict__ fc0w, const float* __restrict__ fc0b,
                            const float* __restrict__ fc1w, const float* __restrict__ fc1b,
                            float* __restrict__ cof) {
  int b = threadIdx.x;
  if (b >= BB) return;
  const float* pv = pooled + b * CC;
  float g[4], nz[4];
  for (int e = 0; e < 4; ++e) {
    float d1 = fc1b[e], d0 = fc0b[e];
    for (int c = 0; c < CC; ++c) { d1 += pv[c]*fc1w[e*CC + c]; d0 += pv[c]*fc0w[e*CC + c]; }
    g[e]  = d1 > 0.f ? d1 : 0.2f * d1;
    nz[e] = (d0 > 20.f) ? d0 : log1pf(__expf(d0));
  }
  float mu = (nz[0]+nz[1]+nz[2]+nz[3]) * 0.25f;
  float ss = 0.f;
  for (int e = 0; e < 4; ++e) { float d = nz[e]-mu; ss += d*d; }
  float sd = sqrtf(ss / 3.f);            // torch std: ddof=1
  if (sd == 0.f) sd = 1.f;
  float h[4];
  for (int e = 0; e < 4; ++e) h[e] = g[e] + (nz[e]-mu)/sd;
  int i1 = 0;
  for (int e = 1; e < 4; ++e) if (h[e] > h[i1]) i1 = e;
  int i2 = -1;
  for (int e = 0; e < 4; ++e) if (e != i1 && (i2 < 0 || h[e] > h[i2])) i2 = e;
  float lg[4], m = fmaxf(g[i1], g[i2]);
  for (int e = 0; e < 4; ++e) lg[e] = (e == i1 || e == i2) ? g[e] : -1e9f;
  float ex[4], den = 0.f;
  for (int e = 0; e < 4; ++e) { ex[e] = __expf(lg[e]-m); den += ex[e]; }
  for (int e = 0; e < 4; ++e) cof[b*4 + e] = ex[e] / den;
}

// ---------------- driver ----------------
extern "C" void kernel_launch(void* const* d_in, const int* in_sizes, int n_in,
                              void* d_out, int out_size, void* d_ws, size_t ws_size,
                              hipStream_t stream) {
  (void)in_sizes; (void)n_in; (void)out_size; (void)ws_size;
  const float* A    = (const float*)d_in[0];
  const float* Bi   = (const float*)d_in[1];
  const float* fc0w = (const float*)d_in[2];
  const float* fc0b = (const float*)d_in[3];
  const float* fc1w = (const float*)d_in[4];
  const float* fc1b = (const float*)d_in[5];
  const float* invl = (const float*)d_in[6];
  const float* invu = (const float*)d_in[7];
  const float* logs = (const float*)d_in[8];
  const float* perm = (const float*)d_in[9];
  const float* sgns = (const float*)d_in[10];
  // 0 = fF, 1 = fG, 2 = fH
  const float *hc1w[3], *hc1b[3], *hgam[3], *hbet[3], *hc2w[3], *hc2b[3];
  for (int t = 0; t < 3; ++t) {
    int base = 11 + 6*t;
    hc1w[t] = (const float*)d_in[base+0]; hc1b[t] = (const float*)d_in[base+1];
    hgam[t] = (const float*)d_in[base+2]; hbet[t] = (const float*)d_in[base+3];
    hc2w[t] = (const float*)d_in[base+4]; hc2b[t] = (const float*)d_in[base+5];
  }
  const float* postw = (const float*)d_in[29];
  const float* postb = (const float*)d_in[30];

  char* ws = (char*)d_ws; size_t off = 0;
  auto alloc = [&](size_t bytes) -> void* {
    void* p = ws + off; off += (bytes + 255) & ~(size_t)255; return p;
  };
  const long NP = (long)BB * HWSZ;     // total pixels

  float*          lu    = (float*)alloc(C2D*C2D*4);
  unsigned short* wbf   = (unsigned short*)alloc(C2D*C2D*2);
  unsigned short* pwbf  = (unsigned short*)alloc(CC*C2D*2);
  unsigned short* wt1[3], *wt2[3];
  for (int t = 0; t < 3; ++t) {
    wt1[t] = (unsigned short*)alloc(CC*9*CC*2);
    wt2[t] = (unsigned short*)alloc(CC*9*CC*2);
  }
  unsigned short* xcat  = (unsigned short*)alloc((size_t)NP*C2D*2);  // NHWC; reused as ycat
  float*          x_f   = (float*)alloc((size_t)NP*C2D*4);           // NHWC x1|x2
  unsigned short* t_in  = (unsigned short*)alloc((size_t)NP*CC*2);   // NHWC bf16
  float*          r_f   = (float*)alloc((size_t)NP*CC*4);            // conv1 out (alias g)
  float*          y1_f  = (float*)alloc((size_t)NP*CC*4);            // alias o_f at end
  float*          s_f   = (float*)alloc((size_t)NP*CC*4);
  float*          part  = (float*)alloc((size_t)BB*16*128*4);
  float*          stats = (float*)alloc(BB*32*2*4);
  float*          pooled= (float*)alloc(BB*CC*4);
  float*          g_f   = r_f;
  unsigned short* ycat  = xcat;
  float*          o_f   = y1_f;        // post-GEMM NHWC output (y1 dead by then)

  build_w_kernel<<<1, 256, 0, stream>>>(invl, invu, logs, perm, sgns, lu, wbf);
  {
    int n3 = CC*9*CC, gb = (n3 + 255)/256;
    for (int t = 0; t < 3; ++t) {
      conv_w_to_bf16<<<gb, 256, 0, stream>>>(hc1w[t], wt1[t]);
      conv_w_to_bf16<<<gb, 256, 0, stream>>>(hc2w[t], wt2[t]);
    }
    mat_interleave_bf16<<<(CC*C2D + 255)/256, 256, 0, stream>>>(postw, pwbf, CC, C2D);
  }
  long n64 = NP * CC;
  int g64 = (int)((n64 + 255)/256);
  int gtile = (int)(NP/64), gconv = BB * HH * 3;

  // a,b (NCHW f32) -> xcat (NHWC bf16 [P][128])
  nchw_to_nhwc_bf16<<<gtile, 256, 0, stream>>>(A,  xcat, 0);
  nchw_to_nhwc_bf16<<<gtile, 256, 0, stream>>>(Bi, xcat, CC);
  gemm1x1_wmma<C2D><<<gtile, 256, 0, stream>>>(xcat, wbf, nullptr, x_f, C2D, 0);

  // --- F block: y1 = x1 + x2 + conv2(HIN(conv1(x2))) ---
  slice_cvt_bf16<<<g64, 256, 0, stream>>>(x_f + CC, C2D, t_in, CC);
  conv3x3_wmma<<<gconv, 256, 0, stream>>>(t_in, CC, wt1[0], hc1b[0],
                                          nullptr, 0, nullptr, 0, r_f, 0);
  hin_stats1<<<BB*16, 256, 0, stream>>>(r_f, part);
  hin_stats2<<<1, 256, 0, stream>>>(part, stats);
  hin_norm_cvt<<<g64, 256, 0, stream>>>(r_f, stats, hgam[0], hbet[0], t_in);
  conv3x3_wmma<<<gconv, 256, 0, stream>>>(t_in, CC, wt2[0], hc2b[0],
                                          x_f + CC, C2D, x_f, C2D, y1_f, 0);
  slice_cvt_bf16<<<g64, 256, 0, stream>>>(y1_f, CC, ycat, C2D);       // ycat[:,0:64]

  // --- H block -> s = 0.8*(2*sigmoid(y1 + conv2(HIN(conv1(y1)))) - 1) ---
  conv3x3_wmma<<<gconv, 256, 0, stream>>>(ycat, C2D, wt1[2], hc1b[2],
                                          nullptr, 0, nullptr, 0, r_f, 0);
  hin_stats1<<<BB*16, 256, 0, stream>>>(r_f, part);
  hin_stats2<<<1, 256, 0, stream>>>(part, stats);
  hin_norm_cvt<<<g64, 256, 0, stream>>>(r_f, stats, hgam[2], hbet[2], t_in);
  conv3x3_wmma<<<gconv, 256, 0, stream>>>(t_in, CC, wt2[2], hc2b[2],
                                          y1_f, CC, nullptr, 0, s_f, 1);

  // --- G block -> g = y1 + conv2(HIN(conv1(y1))) ---
  conv3x3_wmma<<<gconv, 256, 0, stream>>>(ycat, C2D, wt1[1], hc1b[1],
                                          nullptr, 0, nullptr, 0, r_f, 0);
  hin_stats1<<<BB*16, 256, 0, stream>>>(r_f, part);
  hin_stats2<<<1, 256, 0, stream>>>(part, stats);
  hin_norm_cvt<<<g64, 256, 0, stream>>>(r_f, stats, hgam[1], hbet[1], t_in);
  conv3x3_wmma<<<gconv, 256, 0, stream>>>(t_in, CC, wt2[1], hc2b[1],
                                          y1_f, CC, nullptr, 0, g_f, 0);

  // y2 = x2 * exp(s) + g -> ycat[:,64:128]
  y2_kernel<<<g64, 256, 0, stream>>>(x_f + CC, s_f, g_f, ycat + CC);

  // post 1x1 conv (128 -> 64) + leaky(0.01) -> NHWC o_f, then -> NCHW d_out
  float* outx = (float*)d_out;
  gemm1x1_wmma<CC><<<gtile, 256, 0, stream>>>(ycat, pwbf, postb, o_f, CC, 2);
  nhwc_to_nchw_f32<<<gtile, 256, 0, stream>>>(o_f, outx);

  // gate
  pool_kernel<<<BB*CC, 256, 0, stream>>>(outx, pooled);
  gate_kernel<<<1, 32, 0, stream>>>(pooled, fc0w, fc0b, fc1w, fc1b, outx + NP*CC);
}
